// SQ_VAE_36309653520934
// MI455X (gfx1250) — compile-verified, matrix operand order checked by reading, not run on hardware
//
#include <hip/hip_runtime.h>
#include <hip/hip_bf16.h>
#include <math.h>

#define B_ 8
#define T_ 2048
#define D_ 512
#define V_ 2048
#define L_ 3
#define N_ (B_ * T_)                 // 16384 rows
#define KLW_OVER_N 6.103515625e-8f   // 0.001 / 16384
#define LOGV 7.6246189861594f        // log(2048)

typedef __attribute__((ext_vector_type(16))) __bf16 v16bf;
typedef __attribute__((ext_vector_type(8)))  float  v8f;
typedef __attribute__((ext_vector_type(4)))  unsigned int v4u;
typedef __attribute__((ext_vector_type(8)))  int    v8i;
typedef __attribute__((ext_vector_type(4)))  int    v4i;

// Tensor Data Mover availability (arity differs between toolchains)
#if defined(__has_builtin)
#  if __has_builtin(__builtin_amdgcn_tensor_load_to_lds) && \
      __has_builtin(__builtin_amdgcn_s_wait_tensorcnt)
#    define USE_TDM 1
#  endif
#endif
#ifndef USE_TDM
#  define USE_TDM 0
#endif

// ---------------------------------------------------------------------------
// init: curr <- latent ; code_sum <- 0 ; loss <- 0
// ---------------------------------------------------------------------------
__global__ __launch_bounds__(256) void sqvae_init(const float* __restrict__ latent,
                                                  float* __restrict__ curr,
                                                  float* __restrict__ out) {
    size_t i = (size_t)blockIdx.x * 256 + threadIdx.x;
    curr[i] = latent[i];
    out[1 + i] = 0.0f;
    if (i == 0) out[0] = 0.0f;
}

// ---------------------------------------------------------------------------
// per-row prep: scale (std+1e-5 for lvl>0), x_n = l2norm(curr/scale) -> bf16
// ---------------------------------------------------------------------------
__global__ __launch_bounds__(256) void sqvae_prep_rows(const float* __restrict__ curr,
                                                       __hip_bfloat16* __restrict__ xb,
                                                       float* __restrict__ scale_out,
                                                       int lvl) {
    const int lane = threadIdx.x & 31;
    const int row  = blockIdx.x * 8 + (threadIdx.x >> 5);
    const float4* x4 = (const float4*)(curr + (size_t)row * D_);
    float4 v[4];
    float sum = 0.f, sq = 0.f;
#pragma unroll
    for (int i = 0; i < 4; ++i) {
        v[i] = x4[lane + i * 32];
        sum += v[i].x + v[i].y + v[i].z + v[i].w;
        sq  += v[i].x * v[i].x + v[i].y * v[i].y + v[i].z * v[i].z + v[i].w * v[i].w;
    }
#pragma unroll
    for (int off = 1; off < 32; off <<= 1) {
        sum += __shfl_xor(sum, off, 32);
        sq  += __shfl_xor(sq,  off, 32);
    }
    float scale = 1.0f;
    if (lvl > 0) {
        float mean = sum * (1.0f / D_);
        float var  = fmaxf((sq - (float)D_ * mean * mean) * (1.0f / (D_ - 1)), 0.0f);
        scale = sqrtf(var) + 1e-5f;   // torch unbiased std + eps
    }
    float norm_scaled = sqrtf(sq) / scale;             // ||curr/scale||
    float inv = 1.0f / (scale * fmaxf(norm_scaled, 1e-12f));
    if (lane == 0) scale_out[row] = scale;
    __hip_bfloat16* o = xb + (size_t)row * D_;
#pragma unroll
    for (int i = 0; i < 4; ++i) {
        int e = (lane + i * 32) * 4;
        o[e + 0] = __float2bfloat16(v[i].x * inv);
        o[e + 1] = __float2bfloat16(v[i].y * inv);
        o[e + 2] = __float2bfloat16(v[i].z * inv);
        o[e + 3] = __float2bfloat16(v[i].w * inv);
    }
}

// ---------------------------------------------------------------------------
// codebook prep: bw = sigmoid(w) (f32, kept for the gather), e_n -> bf16
// ---------------------------------------------------------------------------
__global__ __launch_bounds__(256) void sqvae_prep_codebook(const float* __restrict__ w,
                                                           float* __restrict__ bw,
                                                           __hip_bfloat16* __restrict__ eb) {
    const int lane = threadIdx.x & 31;
    const int row  = blockIdx.x * 8 + (threadIdx.x >> 5);
    const float4* x4 = (const float4*)(w + (size_t)row * D_);
    float4 v[4];
    float sq = 0.f;
#pragma unroll
    for (int i = 0; i < 4; ++i) {
        float4 t = x4[lane + i * 32];
        t.x = 1.0f / (1.0f + __expf(-t.x));
        t.y = 1.0f / (1.0f + __expf(-t.y));
        t.z = 1.0f / (1.0f + __expf(-t.z));
        t.w = 1.0f / (1.0f + __expf(-t.w));
        v[i] = t;
        sq += t.x * t.x + t.y * t.y + t.z * t.z + t.w * t.w;
    }
#pragma unroll
    for (int off = 1; off < 32; off <<= 1) sq += __shfl_xor(sq, off, 32);
    float inv = 1.0f / fmaxf(sqrtf(sq), 1e-12f);
    float4* bo = (float4*)(bw + (size_t)row * D_);
    __hip_bfloat16* eo = eb + (size_t)row * D_;
#pragma unroll
    for (int i = 0; i < 4; ++i) {
        bo[lane + i * 32] = v[i];
        int e = (lane + i * 32) * 4;
        eo[e + 0] = __float2bfloat16(v[i].x * inv);
        eo[e + 1] = __float2bfloat16(v[i].y * inv);
        eo[e + 2] = __float2bfloat16(v[i].z * inv);
        eo[e + 3] = __float2bfloat16(v[i].w * inv);
    }
}

// ---------------------------------------------------------------------------
// Fused GEMM (bf16 WMMA, f32 acc) + online softmax / argmax(logits+gumbel) / KL
// 4 waves / block; each wave owns 16 rows (A tile resident in 128 VGPRs).
// B tiles (16 codes x 512 K = 16KB) are double-buffered through LDS.
// TDM path: one tensor_load_to_lds per tile (wave 0), D# pad fields reproduce
// the 1024B-row + 16B-pad layout so the 16 lanes' ds_load_b128 hit all 64 banks.
// Fallback path: cooperative global->reg->LDS staging.
// ---------------------------------------------------------------------------
union FragU { v16bf v; uint4 q[2]; };

#define LDS_ROW_STRIDE 1040          // 1024B data + 16B pad
#define LDS_BUF_BYTES (16 * LDS_ROW_STRIDE)

#if USE_TDM
// Issue a TDM load of one 16-row x 1024B tile (row-contiguous in memory) into
// LDS at lds_off with 16B padding after every 1024B row.  ISA 8.3-8.6 layout.
__device__ __forceinline__ void tdm_load_tile(unsigned long long ga, unsigned lds_off) {
    v4u g0;
    g0.x = 1u;                                              // count=1 (user D#)
    g0.y = lds_off;                                         // lds_addr (bytes)
    g0.z = (unsigned)(ga & 0xffffffffu);                    // global_addr[31:0]
    g0.w = (unsigned)((ga >> 32) & 0x01ffffffu) | (2u << 30); // addr[56:32] | type=2
    v8i g1;
    // data_size=3 (8B units), pad_enable, pad_interval=7 (256 DW = 1024B),
    // pad_amount=3 (4 DW = 16B)
    g1[0] = (int)((3u << 16) | (1u << 20) | (7u << 22) | (3u << 25));
    g1[1] = (int)(128u << 16);   // tensor_dim0[15:0] = 128 (8B units per row)
    g1[2] = (int)(16u << 16);    // tensor_dim0 hi=0 ; tensor_dim1[15:0] = 16 rows
    g1[3] = (int)(128u << 16);   // tensor_dim1 hi=0 ; tile_dim0 = 128
    g1[4] = (int)16u;            // tile_dim1 = 16 ; tile_dim2 = 0
    g1[5] = (int)128u;           // tensor_dim0_stride = 128 (contiguous rows)
    g1[6] = 0;                   // stride hi / tensor_dim1_stride lo
    g1[7] = 0;
    v4i gz = {0, 0, 0, 0};
#if __clang_major__ >= 23
    v8i gz8 = {0, 0, 0, 0, 0, 0, 0, 0};
    __builtin_amdgcn_tensor_load_to_lds(g0, g1, gz, gz, gz8, 0);
#else
    __builtin_amdgcn_tensor_load_to_lds(g0, g1, gz, gz, 0);
#endif
}
#endif

__global__ __launch_bounds__(128, 1) void sqvae_gemm_softmax(
        const __hip_bfloat16* __restrict__ xb,   // [N, D] bf16, l2-normed rows
        const __hip_bfloat16* __restrict__ eb,   // [V, D] bf16, l2-normed codes
        const float* __restrict__ gumbel,        // [N, V] (level slice)
        const float* __restrict__ logit_scales,  // [L]
        int lvl,
        int* __restrict__ idx_out,               // [N]
        float* __restrict__ loss_out) {          // &d_out[0]
    __shared__ __align__(16) char lds[2 * LDS_BUF_BYTES];

    const int tid     = threadIdx.x;
    const int lane    = tid & 31;
    const int wave    = tid >> 5;
    const int rowTile = blockIdx.x * 4 + wave;   // 0..N/16-1
    const int rowBase = rowTile * 16;
    const int sel     = lane >> 4;               // half-wave select (K swizzle)
    const int nsub    = lane & 15;
    const float ls    = logit_scales[lvl];
    const char* ebase = (const char*)eb;

    // ---- load A fragments: row = rowBase + nsub, ISA 16-bit A swizzle ----
    v16bf a[16];
    const char* arow = (const char*)(xb + (size_t)(rowBase + nsub) * D_);
#pragma unroll
    for (int k = 0; k < 16; ++k) {
        FragU u;
        u.q[0] = *(const uint4*)(arow + k * 64 + sel * 16);        // K 0..7 / 8..15
        u.q[1] = *(const uint4*)(arow + k * 64 + 32 + sel * 16);   // K 16..23 / 24..31
        a[k] = u.v;
    }

    float m[8], s[8], t[8], amax[8];
    int aidx[8];
#pragma unroll
    for (int r = 0; r < 8; ++r) {
        m[r] = -3.0e38f; s[r] = 0.f; t[r] = 0.f; amax[r] = -3.0e38f; aidx[r] = 0;
    }

#if USE_TDM
    const unsigned lds_base = (unsigned)(uintptr_t)(void*)lds;
    if (wave == 0) {
        tdm_load_tile((unsigned long long)(uintptr_t)ebase, lds_base);
        __builtin_amdgcn_s_wait_tensorcnt(0);
    }
    __syncthreads();
#else
    // cooperative chunk mapping: 16KB tile = 1024 x 16B chunks, 8 per thread
    uint4 pf[8];
#pragma unroll
    for (int i = 0; i < 8; ++i) {
        int c = tid + i * 128;
        int row = c >> 6, off = (c & 63) << 4;
        pf[i] = *(const uint4*)(ebase + (size_t)row * 1024 + off);
    }
#pragma unroll
    for (int i = 0; i < 8; ++i) {
        int c = tid + i * 128;
        int row = c >> 6, off = (c & 63) << 4;
        *(uint4*)(lds + row * LDS_ROW_STRIDE + off) = pf[i];
    }
    __syncthreads();
#endif

    for (int tile = 0; tile < V_ / 16; ++tile) {
        const char* cur = lds + (tile & 1) * LDS_BUF_BYTES;

#if USE_TDM
        // next tile's buffer was last READ in iteration tile-1 (pre-barrier),
        // so the DMA write here is race-free with one barrier per iteration.
        if (wave == 0 && tile + 1 < V_ / 16) {
            tdm_load_tile((unsigned long long)(uintptr_t)(ebase + (size_t)(tile + 1) * 16 * 1024),
                          lds_base + (unsigned)(((tile + 1) & 1) * LDS_BUF_BYTES));
        }
#else
        char* nxt = lds + ((tile + 1) & 1) * LDS_BUF_BYTES;
        uint4 pf[8];
        if (tile + 1 < V_ / 16) {
#pragma unroll
            for (int i = 0; i < 8; ++i) {
                int c = tid + i * 128;
                int row = c >> 6, off = (c & 63) << 4;
                pf[i] = *(const uint4*)(ebase + ((size_t)(tile + 1) * 16 + row) * 1024 + off);
            }
        }
#endif

        // ---- 16x16 logits tile: 16 WMMAs over K=512, B from LDS ----
        v8f cacc = {};
        const char* brow = cur + nsub * LDS_ROW_STRIDE;
#pragma unroll
        for (int k = 0; k < 16; ++k) {
            FragU u;
            u.q[0] = *(const uint4*)(brow + k * 64 + sel * 16);
            u.q[1] = *(const uint4*)(brow + k * 64 + 32 + sel * 16);
            cacc = __builtin_amdgcn_wmma_f32_16x16x32_bf16(false, a[k], false, u.v,
                                                           (short)0, cacc, false, false);
        }

        // C layout: element r in lane L = C[M, N], M = r + 8*sel, N = nsub
        const int col = tile * 16 + nsub;
#pragma unroll
        for (int r = 0; r < 8; ++r) {
            const int M = r + sel * 8;
            float logit = cacc[r] * ls;                  // TAU == 1
            float g  = gumbel[(size_t)(rowBase + M) * V_ + col];
            float av = logit + g;
            bool take = (av > amax[r]);
            amax[r] = take ? av : amax[r];
            aidx[r] = take ? col : aidx[r];
            // branchless online softmax (m, sum e^{l-m}, sum l*e^{l-m})
            float mn = fmaxf(m[r], logit);
            float f0 = __expf(m[r] - mn);
            float e  = __expf(logit - mn);
            s[r] = s[r] * f0 + e;
            t[r] = t[r] * f0 + logit * e;
            m[r] = mn;
        }

#if USE_TDM
        if (wave == 0) __builtin_amdgcn_s_wait_tensorcnt(0);
        __syncthreads();                 // DMA'd tile visible to all waves
#else
        __syncthreads();                 // all waves done reading sibling buffer
        if (tile + 1 < V_ / 16) {
#pragma unroll
            for (int i = 0; i < 8; ++i) {
                int c = tid + i * 128;
                int row = c >> 6, off = (c & 63) << 4;
                *(uint4*)(nxt + row * LDS_ROW_STRIDE + off) = pf[i];
            }
        }
        __syncthreads();                 // next tile visible to all waves
#endif
    }

    // ---- merge across the 16 lanes holding the same 8 rows ----
#pragma unroll
    for (int off = 1; off < 16; off <<= 1) {
#pragma unroll
        for (int r = 0; r < 8; ++r) {
            float om = __shfl_xor(m[r], off, 16);
            float os = __shfl_xor(s[r], off, 16);
            float ot = __shfl_xor(t[r], off, 16);
            float mm = fmaxf(m[r], om);
            float f1 = __expf(m[r] - mm);
            float f2 = __expf(om - mm);
            s[r] = s[r] * f1 + os * f2;
            t[r] = t[r] * f1 + ot * f2;
            m[r] = mm;
            float oa = __shfl_xor(amax[r], off, 16);
            int   oi = __shfl_xor(aidx[r], off, 16);
            if (oa > amax[r] || (oa == amax[r] && oi < aidx[r])) {
                amax[r] = oa; aidx[r] = oi;
            }
        }
    }

    if (nsub == 0) {
        float klsum = 0.f;
#pragma unroll
        for (int r = 0; r < 8; ++r) {
            // sum p*logp = t/s - lse ; kl = that + logV
            float kl = t[r] / s[r] - (m[r] + __logf(s[r])) + LOGV;
            klsum += kl;
            idx_out[rowBase + r + sel * 8] = aidx[r];
        }
        atomicAdd(loss_out, klsum * KLW_OVER_N);
    }
}

// ---------------------------------------------------------------------------
// update: q = bw[idx[n]] * scale[n] ; code_sum += q ; curr -= q
// ---------------------------------------------------------------------------
__global__ __launch_bounds__(256) void sqvae_update(const float* __restrict__ bw,
                                                    const int* __restrict__ idx,
                                                    const float* __restrict__ scale,
                                                    float* __restrict__ curr,
                                                    float* __restrict__ code_sum) {
    size_t i = (size_t)blockIdx.x * 256 + threadIdx.x;
    int n = (int)(i >> 9);          // / D_
    int d = (int)(i & (D_ - 1));
    float q = bw[(size_t)idx[n] * D_ + d] * scale[n];
    code_sum[i] += q;
    curr[i] -= q;
}

// ---------------------------------------------------------------------------
extern "C" void kernel_launch(void* const* d_in, const int* in_sizes, int n_in,
                              void* d_out, int out_size, void* d_ws, size_t ws_size,
                              hipStream_t stream) {
    const float* latent       = (const float*)d_in[0];   // [B,T,D]
    const float* gumbel       = (const float*)d_in[1];   // [L,N,V]
    const float* embed_w      = (const float*)d_in[2];   // [L,V,D]
    const float* logit_scales = (const float*)d_in[3];   // [L]
    float* out = (float*)d_out;                          // [0]=loss, [1..]=code_sum

    const size_t ND = (size_t)N_ * D_;   // 8388608
    const size_t VD = (size_t)V_ * D_;   // 1048576

    char* p = (char*)d_ws;
    float* curr = (float*)p;           p += ND * sizeof(float);
    float* bw   = (float*)p;           p += VD * sizeof(float);
    float* scl  = (float*)p;           p += (size_t)N_ * sizeof(float);
    int*   idx  = (int*)p;             p += (size_t)N_ * sizeof(int);
    __hip_bfloat16* xb = (__hip_bfloat16*)p;  p += ND * sizeof(__hip_bfloat16);
    __hip_bfloat16* eb = (__hip_bfloat16*)p;  p += VD * sizeof(__hip_bfloat16);

    sqvae_init<<<(int)(ND / 256), 256, 0, stream>>>(latent, curr, out);

    for (int lvl = 0; lvl < L_; ++lvl) {
        sqvae_prep_rows<<<N_ / 8, 256, 0, stream>>>(curr, xb, scl, lvl);
        sqvae_prep_codebook<<<V_ / 8, 256, 0, stream>>>(embed_w + (size_t)lvl * VD, bw, eb);
        sqvae_gemm_softmax<<<(N_ / 16) / 4, 128, 0, stream>>>(
            xb, eb, gumbel + (size_t)lvl * N_ * V_, logit_scales, lvl, idx, out);
        sqvae_update<<<(int)(ND / 256), 256, 0, stream>>>(bw, idx, scl, curr, out + 1);
    }
}
